// AgentLayerC_v2_13623636263378
// MI455X (gfx1250) — compile-verified, matrix operand order checked by reading, not run on hardware
//
#include <hip/hip_runtime.h>
#include <hip/hip_bf16.h>
#include <math.h>

typedef __attribute__((ext_vector_type(16))) _Float16 v16h;
typedef __attribute__((ext_vector_type(8)))  float    v8f;

// ---- problem constants ----
static constexpr int BSEG = 64;
static constexpr int PTS  = 4096;
static constexpr int NPTS = BSEG * PTS;     // 262144
static constexpr int OBSD = 128;
#define LOG2PIE 2.8378770664093453f

// ---- workspace layout (float offsets; all 16B-aligned) ----
static constexpr int OFF_OBS_SUM = 0;        // 8192 (becomes obs_agg after divide)
static constexpr int OFF_OBS_ATT = 8192;     // 8192
static constexpr int OFF_DENOM   = 16384;    // 64
static constexpr int OFF_SEGMAX  = 16448;    // 64
static constexpr int OFF_UNC     = 16512;    // 64
static constexpr int OFF_X       = 16576;    // 64*96   (x padded 70 -> 96)
static constexpr int OFF_WIH     = 22720;    // 768*96  (W_ih padded 70 -> 96)
static constexpr int OFF_GI      = 96448;    // 64*768
static constexpr int OFF_GH      = 145600;   // 64*768
static constexpr int OFF_POST_IN = 194752;   // 64*384
static constexpr int OFF_POST    = 219328;   // 64*128
static constexpr int OFF_AE1     = 227520;   // 64*256
static constexpr int OFF_ENC     = 243904;   // 64*128
static constexpr int OFF_CTX_IN  = 252096;   // 64*448
static constexpr int OFF_C1      = 280768;   // 64*512

// ---- output layout (float offsets in d_out) ----
static constexpr int OUT_H_OFF   = 0;        // 64*256
static constexpr int OUT_Z_OFF   = 16384;    // 64*64
static constexpr int OUT_CTX_OFF = 20480;    // 64*128
static constexpr int OUT_PRI_OFF = 28672;    // 262144
static constexpr int OUT_PN_OFF  = 290816;   // 262144
static constexpr int OUT_UNC_OFF = 552960;   // 64

// ===================================================================
// init: zero the two atomic accumulators (obs_sum + obs_att)
// ===================================================================
__global__ void k_init(float* __restrict__ ws) {
    int i = blockIdx.x * blockDim.x + threadIdx.x;
    if (i < 16384) ws[i] = 0.0f;
}

// ===================================================================
// streaming pass 1: per-segment sum of obs (segments are contiguous)
// grid (64 segs, 32 chunks of 128 rows), block 128.
// lane = 4-column group; wave reads a full 512B row with b128 loads.
// ===================================================================
__global__ void k_segsum(const float* __restrict__ obs, float* __restrict__ obs_sum) {
    const int seg = blockIdx.x, chunk = blockIdx.y;
    const int quad = threadIdx.x & 31;   // column group (4 cols)
    const int rg   = threadIdx.x >> 5;   // row phase 0..3
    const float4* p = (const float4*)(obs + (size_t)(seg * PTS + chunk * 128) * OBSD) + quad;
    float4 acc = {0.f, 0.f, 0.f, 0.f};
#pragma unroll 4
    for (int r = rg; r < 128; r += 4) {
        __builtin_prefetch(p + (size_t)(r + 16) * 32, 0, 1);  // global_prefetch_b8
        float4 v = p[(size_t)r * 32];
        acc.x += v.x; acc.y += v.y; acc.z += v.z; acc.w += v.w;
    }
    float* dst = &obs_sum[seg * OBSD + quad * 4];
    atomicAdd(dst + 0, acc.x); atomicAdd(dst + 1, acc.y);
    atomicAdd(dst + 2, acc.z); atomicAdd(dst + 3, acc.w);
}

// obs_agg = obs_sum / PTS   (counts are exactly 4096)
__global__ void k_agg_div(float* __restrict__ s) {
    int i = blockIdx.x * blockDim.x + threadIdx.x;
    if (i < BSEG * OBSD) s[i] *= (1.0f / (float)PTS);
}

// ===================================================================
// generic tiny GEMM: OUT(64xN) = act( X(64xK) @ W(NxK)^T + bias )
// one wave per 16x16 tile, v_wmma_f32_16x16x32_f16, f32 accumulate.
// Fragment layouts (ISA 7.12.2):
//   A 16x32 f16: lane lo = M row; elems 0..7 -> K = base..base+7,
//   elems 8..15 -> K = base+16..base+23, base = kc + 8*hi. B mirrored
//   with lane = N column. D: N = lane&15, M = vgpr + 8*hi.
// K, N compile-time; K % 32 == 0 -> branch-free b128 fragment loads.
// ===================================================================
template <int ACT, int K, int N>  // ACT: 0 = identity, 1 = relu
__global__ void k_gemm_wmma(const float* __restrict__ X, const float* __restrict__ W,
                            const float* __restrict__ bias, float* __restrict__ OUT) {
    static_assert(K % 32 == 0, "K must be a multiple of 32");
    const int lane = threadIdx.x;
    const int lo = lane & 15;
    const int hi = lane >> 4;
    const int bm = blockIdx.y * 16;
    const int bn = blockIdx.x * 16;
    const float* xrow = X + (bm + lo) * K;
    const float* wrow = W + (bn + lo) * K;

    v8f c = {};
#pragma unroll
    for (int kc = 0; kc < K; kc += 32) {
        const int base = kc + hi * 8;
        float4 a0 = *(const float4*)(xrow + base);
        float4 a1 = *(const float4*)(xrow + base + 4);
        float4 a2 = *(const float4*)(xrow + base + 16);
        float4 a3 = *(const float4*)(xrow + base + 20);
        float4 b0 = *(const float4*)(wrow + base);
        float4 b1 = *(const float4*)(wrow + base + 4);
        float4 b2 = *(const float4*)(wrow + base + 16);
        float4 b3 = *(const float4*)(wrow + base + 20);
        v16h a, b;
        a[0]  = (_Float16)a0.x; a[1]  = (_Float16)a0.y; a[2]  = (_Float16)a0.z; a[3]  = (_Float16)a0.w;
        a[4]  = (_Float16)a1.x; a[5]  = (_Float16)a1.y; a[6]  = (_Float16)a1.z; a[7]  = (_Float16)a1.w;
        a[8]  = (_Float16)a2.x; a[9]  = (_Float16)a2.y; a[10] = (_Float16)a2.z; a[11] = (_Float16)a2.w;
        a[12] = (_Float16)a3.x; a[13] = (_Float16)a3.y; a[14] = (_Float16)a3.z; a[15] = (_Float16)a3.w;
        b[0]  = (_Float16)b0.x; b[1]  = (_Float16)b0.y; b[2]  = (_Float16)b0.z; b[3]  = (_Float16)b0.w;
        b[4]  = (_Float16)b1.x; b[5]  = (_Float16)b1.y; b[6]  = (_Float16)b1.z; b[7]  = (_Float16)b1.w;
        b[8]  = (_Float16)b2.x; b[9]  = (_Float16)b2.y; b[10] = (_Float16)b2.z; b[11] = (_Float16)b2.w;
        b[12] = (_Float16)b3.x; b[13] = (_Float16)b3.y; b[14] = (_Float16)b3.z; b[15] = (_Float16)b3.w;
        c = __builtin_amdgcn_wmma_f32_16x16x32_f16(false, a, false, b, (short)0, c,
                                                   false, false);
    }
#pragma unroll
    for (int r = 0; r < 8; ++r) {
        int m = bm + r + hi * 8;
        int n = bn + lo;
        float v = c[r] + bias[n];
        if (ACT == 1) v = v > 0.0f ? v : 0.0f;
        OUT[m * N + n] = v;
    }
}

// ===================================================================
// small prep / elementwise kernels
// ===================================================================
// x (64 x 96): [z_prev(64) | action(5) | coh(1) | zero pad(26)]
__global__ void k_prep_x(const float* __restrict__ z_prev, const float* __restrict__ action,
                         const float* __restrict__ coh, float* __restrict__ x) {
    int b = blockIdx.x, t = threadIdx.x;  // block 96
    float v;
    if (t < 64)       v = z_prev[b * 64 + t];
    else if (t < 69)  v = action[b * 5 + (t - 64)];
    else if (t == 69) v = coh[b];
    else              v = 0.0f;
    x[b * 96 + t] = v;
}

// W_ih (768 x 70) -> padded (768 x 96)
__global__ void k_pad_wih(const float* __restrict__ w, float* __restrict__ wp) {
    int i = blockIdx.x * 256 + threadIdx.x;  // 768*96 = 73728
    if (i >= 768 * 96) return;
    int row = i / 96, col = i - row * 96;
    wp[i] = (col < 70) ? w[row * 70 + col] : 0.0f;
}

__global__ void k_gru(const float* __restrict__ gi, const float* __restrict__ gh,
                      const float* __restrict__ h_prev, float* __restrict__ h_out) {
    int i = blockIdx.x * 256 + threadIdx.x;  // 16384 total
    int b = i >> 8, j = i & 255;
    float rr = gi[b * 768 + j] + gh[b * 768 + j];
    float uu = gi[b * 768 + 256 + j] + gh[b * 768 + 256 + j];
    float ni = gi[b * 768 + 512 + j];
    float nh = gh[b * 768 + 512 + j];
    float r = 1.0f / (1.0f + expf(-rr));
    float u = 1.0f / (1.0f + expf(-uu));
    float n = tanhf(ni + r * nh);
    h_out[i] = (1.0f - u) * n + u * h_prev[i];
}

__global__ void k_prep_post(const float* __restrict__ h, const float* __restrict__ agg,
                            float* __restrict__ out) {
    int b = blockIdx.x, t = threadIdx.x;  // block 384
    out[b * 384 + t] = (t < 256) ? h[b * 256 + t] : agg[b * 128 + (t - 256)];
}

// z = mu_q; uncertainty = 0.5 * (sum(logvar_q) + LAT*log(2*pi*e))
__global__ void k_z_unc(const float* __restrict__ post, float* __restrict__ z_out,
                        float* __restrict__ unc_ws, float* __restrict__ unc_out) {
    __shared__ float sm[64];
    int b = blockIdx.x, j = threadIdx.x;  // block 64
    z_out[b * 64 + j] = post[b * 128 + j];
    sm[j] = post[b * 128 + 64 + j];
    __syncthreads();
    for (int s = 32; s > 0; s >>= 1) {
        if (j < s) sm[j] += sm[j + s];
        __syncthreads();
    }
    if (j == 0) {
        float u = 0.5f * (sm[0] + 64.0f * LOG2PIE);
        unc_ws[b] = u;
        unc_out[b] = u;
    }
}

// ===================================================================
// per-segment softmax (segments contiguous -> block reductions)
// ===================================================================
__global__ void k_pri_max(const float* __restrict__ coh_sp, const float* __restrict__ unc,
                          float* __restrict__ pri_out, float* __restrict__ segmax) {
    __shared__ float sm[256];
    int seg = blockIdx.x, t = threadIdx.x;
    float u = unc[seg];
    float mx = -INFINITY;
    for (int i = t; i < PTS; i += 256) {
        float p = coh_sp[seg * PTS + i] * u;   // TEMP = 1
        pri_out[seg * PTS + i] = p;
        mx = fmaxf(mx, p);
    }
    sm[t] = mx;
    __syncthreads();
    for (int s = 128; s > 0; s >>= 1) {
        if (t < s) sm[t] = fmaxf(sm[t], sm[t + s]);
        __syncthreads();
    }
    if (t == 0) segmax[seg] = sm[0];
}

__global__ void k_pri_exp(const float* __restrict__ pri, const float* __restrict__ segmax,
                          float* __restrict__ pn_out, float* __restrict__ denom) {
    __shared__ float sm[256];
    int seg = blockIdx.x, t = threadIdx.x;
    float m = segmax[seg];
    float acc = 0.0f;
    for (int i = t; i < PTS; i += 256) {
        float e = expf(pri[seg * PTS + i] - m);
        pn_out[seg * PTS + i] = e;
        acc += e;
    }
    sm[t] = acc;
    __syncthreads();
    for (int s = 128; s > 0; s >>= 1) {
        if (t < s) sm[t] += sm[t + s];
        __syncthreads();
    }
    if (t == 0) denom[seg] = sm[0];
}

__global__ void k_pri_norm(float* __restrict__ pn, const float* __restrict__ denom) {
    int i = blockIdx.x * 256 + threadIdx.x;  // 262144 total
    int seg = i >> 12;
    pn[i] = pn[i] / fmaxf(denom[seg], 1e-12f);
}

// ===================================================================
// streaming pass 2: priority-weighted per-segment sum of obs
// ===================================================================
__global__ void k_attend(const float* __restrict__ obs, const float* __restrict__ pn,
                         float* __restrict__ att) {
    const int seg = blockIdx.x, chunk = blockIdx.y;
    const int quad = threadIdx.x & 31;
    const int rg   = threadIdx.x >> 5;
    const size_t base = (size_t)(seg * PTS + chunk * 128);
    const float4* p = (const float4*)(obs + base * OBSD) + quad;
    const float* w = pn + base;
    float4 acc = {0.f, 0.f, 0.f, 0.f};
#pragma unroll 4
    for (int r = rg; r < 128; r += 4) {
        __builtin_prefetch(p + (size_t)(r + 16) * 32, 0, 1);
        float4 v = p[(size_t)r * 32];
        float s = w[r];
        acc.x += v.x * s; acc.y += v.y * s; acc.z += v.z * s; acc.w += v.w * s;
    }
    float* dst = &att[seg * OBSD + quad * 4];
    atomicAdd(dst + 0, acc.x); atomicAdd(dst + 1, acc.y);
    atomicAdd(dst + 2, acc.z); atomicAdd(dst + 3, acc.w);
}

__global__ void k_prep_ctx(const float* __restrict__ h, const float* __restrict__ z,
                           const float* __restrict__ enc, float* __restrict__ out) {
    int b = blockIdx.x, t = threadIdx.x;  // block 448
    float v;
    if (t < 256)      v = h[b * 256 + t];
    else if (t < 320) v = z[b * 64 + (t - 256)];
    else              v = enc[b * 128 + (t - 320)];
    out[b * 448 + t] = v;
}

// ===================================================================
extern "C" void kernel_launch(void* const* d_in, const int* in_sizes, int n_in,
                              void* d_out, int out_size, void* d_ws, size_t ws_size,
                              hipStream_t stream) {
    (void)in_sizes; (void)n_in; (void)out_size; (void)ws_size;

    const float* obs        = (const float*)d_in[0];
    const float* action     = (const float*)d_in[1];
    const float* coh_scalar = (const float*)d_in[2];
    const float* coh_sp     = (const float*)d_in[3];
    const float* h_prev     = (const float*)d_in[4];
    const float* z_prev     = (const float*)d_in[5];
    // d_in[6] = batch (unused: segments are contiguous, seg = i / PTS)
    const float* W_ih   = (const float*)d_in[7];
    const float* b_ih   = (const float*)d_in[8];
    const float* W_hh   = (const float*)d_in[9];
    const float* b_hh   = (const float*)d_in[10];
    // W_prior / b_prior (11,12) are dead code: prior stats never returned
    const float* W_post = (const float*)d_in[13];
    const float* b_post = (const float*)d_in[14];
    const float* W_ae1  = (const float*)d_in[15];
    const float* b_ae1  = (const float*)d_in[16];
    const float* W_ae2  = (const float*)d_in[17];
    const float* b_ae2  = (const float*)d_in[18];
    const float* W_c1   = (const float*)d_in[19];
    const float* b_c1   = (const float*)d_in[20];
    const float* W_c2   = (const float*)d_in[21];
    const float* b_c2   = (const float*)d_in[22];

    float* ws  = (float*)d_ws;
    float* out = (float*)d_out;
    float* OUT_H   = out + OUT_H_OFF;
    float* OUT_Z   = out + OUT_Z_OFF;
    float* OUT_CTX = out + OUT_CTX_OFF;
    float* OUT_PRI = out + OUT_PRI_OFF;
    float* OUT_PN  = out + OUT_PN_OFF;
    float* OUT_UNC = out + OUT_UNC_OFF;

    // 0) zero atomic accumulators; pad W_ih (independent of pass 1)
    k_init<<<128, 128, 0, stream>>>(ws);
    k_pad_wih<<<288, 256, 0, stream>>>(W_ih, ws + OFF_WIH);

    // 1) big pass 1: obs segment sums -> mean
    k_segsum<<<dim3(BSEG, 32), 128, 0, stream>>>(obs, ws + OFF_OBS_SUM);
    k_agg_div<<<64, 128, 0, stream>>>(ws + OFF_OBS_SUM);

    // 2) GRU cell (WMMA for both projections)
    k_prep_x<<<BSEG, 96, 0, stream>>>(z_prev, action, coh_scalar, ws + OFF_X);
    k_gemm_wmma<0, 96, 768><<<dim3(48, 4), 32, 0, stream>>>(ws + OFF_X, ws + OFF_WIH,
                                                            b_ih, ws + OFF_GI);
    k_gemm_wmma<0, 256, 768><<<dim3(48, 4), 32, 0, stream>>>(h_prev, W_hh,
                                                             b_hh, ws + OFF_GH);
    k_gru<<<64, 256, 0, stream>>>(ws + OFF_GI, ws + OFF_GH, h_prev, OUT_H);

    // 3) posterior -> z, uncertainty
    k_prep_post<<<BSEG, 384, 0, stream>>>(OUT_H, ws + OFF_OBS_SUM, ws + OFF_POST_IN);
    k_gemm_wmma<0, 384, 128><<<dim3(8, 4), 32, 0, stream>>>(ws + OFF_POST_IN, W_post,
                                                            b_post, ws + OFF_POST);
    k_z_unc<<<BSEG, 64, 0, stream>>>(ws + OFF_POST, OUT_Z, ws + OFF_UNC, OUT_UNC);

    // 4) per-segment softmax over priorities
    k_pri_max<<<BSEG, 256, 0, stream>>>(coh_sp, ws + OFF_UNC, OUT_PRI, ws + OFF_SEGMAX);
    k_pri_exp<<<BSEG, 256, 0, stream>>>(OUT_PRI, ws + OFF_SEGMAX, OUT_PN, ws + OFF_DENOM);
    k_pri_norm<<<NPTS / 256, 256, 0, stream>>>(OUT_PN, ws + OFF_DENOM);

    // 5) big pass 2: attended aggregation (obs should still be L2-resident)
    k_attend<<<dim3(BSEG, 32), 128, 0, stream>>>(obs, OUT_PN, ws + OFF_OBS_ATT);

    // 6) autoencoder + context head (WMMA)
    k_gemm_wmma<1, 128, 256><<<dim3(16, 4), 32, 0, stream>>>(ws + OFF_OBS_ATT, W_ae1,
                                                             b_ae1, ws + OFF_AE1);
    k_gemm_wmma<0, 256, 128><<<dim3(8, 4), 32, 0, stream>>>(ws + OFF_AE1, W_ae2,
                                                            b_ae2, ws + OFF_ENC);
    k_prep_ctx<<<BSEG, 448, 0, stream>>>(OUT_H, OUT_Z, ws + OFF_ENC, ws + OFF_CTX_IN);
    k_gemm_wmma<1, 448, 512><<<dim3(32, 4), 32, 0, stream>>>(ws + OFF_CTX_IN, W_c1,
                                                             b_c1, ws + OFF_C1);
    k_gemm_wmma<0, 512, 128><<<dim3(8, 4), 32, 0, stream>>>(ws + OFF_C1, W_c2,
                                                            b_c2, OUT_CTX);
}
